// HashCodingLayer_9887014716272
// MI455X (gfx1250) — 16-bit and 8-bit matrix operand order repaired and checked by reading, compiled, not run on hardware
//
#include <hip/hip_runtime.h>

typedef __attribute__((ext_vector_type(16))) __bf16 v16bf;
typedef __attribute__((ext_vector_type(8)))  __bf16 v8bf;
typedef __attribute__((ext_vector_type(8)))  float  v8f;
typedef __attribute__((ext_vector_type(8)))  int    v8i;
typedef __attribute__((ext_vector_type(4)))  int    v4i;

#define FEAT    4096
#define HASH    128
#define KCHUNK  64
#define NCHUNK  (FEAT / KCHUNK)
#define TILE_M  128
#define KPAD    80

__device__ __forceinline__ void async_b128(void* lds, const void* g) {
#if __has_builtin(__builtin_amdgcn_global_load_async_to_lds_b128)
    __builtin_amdgcn_global_load_async_to_lds_b128(
        (__attribute__((address_space(1))) v4i*)g,
        (__attribute__((address_space(3))) v4i*)lds, 0, 0);
#else
    *(int4*)lds = *(const int4*)g;
#endif
}

__device__ __forceinline__ void wait_async0() {
#if __has_builtin(__builtin_amdgcn_s_wait_asynccnt)
    __builtin_amdgcn_s_wait_asynccnt(0);
#else
    asm volatile("s_wait_asynccnt 0" ::: "memory");
#endif
}

__global__ __launch_bounds__(256)
void convert_w_kernel(const float* __restrict__ W, __bf16* __restrict__ Wbf)
{
    const int i = (blockIdx.x * 256 + threadIdx.x) * 8;
    float4 f0 = *(const float4*)(W + i);
    float4 f1 = *(const float4*)(W + i + 4);
    v8bf h;
    h[0] = (__bf16)f0.x; h[1] = (__bf16)f0.y; h[2] = (__bf16)f0.z; h[3] = (__bf16)f0.w;
    h[4] = (__bf16)f1.x; h[5] = (__bf16)f1.y; h[6] = (__bf16)f1.z; h[7] = (__bf16)f1.w;
    *(v8bf*)(Wbf + i) = h;
}

__global__ __launch_bounds__(256)
void binarize_wmma_kernel(const float*  __restrict__ X,
                          const __bf16* __restrict__ Wbf,
                          const float*  __restrict__ bias,
                          signed char*  __restrict__ out,
                          int rows)
{
    __shared__ __bf16 Bt[2][HASH][KPAD];

    const int tid   = threadIdx.x;
    const int wave  = tid >> 5;
    const int lane  = tid & 31;
    const int lhalf = lane >> 4;
    const int l15   = lane & 15;
    const int mbase = blockIdx.x * TILE_M + wave * 16;

    v8f acc[8] = {};

    int arow = mbase + l15;
    if (arow >= rows) arow = rows - 1;
    const float* aptr = X + (size_t)arow * FEAT;

    const int sn = tid >> 1;
    const int sq = (tid & 1) * 32;

    {
        const __bf16* g = Wbf + (size_t)sn * FEAT + sq;
        __bf16*       l = &Bt[0][sn][sq];
        async_b128(l, g); async_b128(l + 8, g + 8);
        async_b128(l + 16, g + 16); async_b128(l + 24, g + 24);
    }

    for (int c = 0; c < NCHUNK; ++c) {
        const int kb = c * KCHUNK;
        wait_async0();
        __syncthreads();

        if (c + 1 < NCHUNK) {
            const __bf16* g = Wbf + (size_t)sn * FEAT + (kb + KCHUNK) + sq;
            __bf16*       l = &Bt[(c + 1) & 1][sn][sq];
            async_b128(l, g); async_b128(l + 8, g + 8);
            async_b128(l + 16, g + 16); async_b128(l + 24, g + 24);
        }

        if (kb + KCHUNK < FEAT)
            __builtin_prefetch(aptr + kb + KCHUNK, 0, 1);

        const int klo = lhalf * 16;
        float4 fA[8];
        #pragma unroll
        for (int s = 0; s < 2; ++s) {
            const int base = kb + s * 32;
            fA[s * 4 + 0] = *(const float4*)(aptr + base + klo);
            fA[s * 4 + 1] = *(const float4*)(aptr + base + klo + 4);
            fA[s * 4 + 2] = *(const float4*)(aptr + base + klo + 8);
            fA[s * 4 + 3] = *(const float4*)(aptr + base + klo + 12);
        }

        const int buf = c & 1;
        #pragma unroll
        for (int s = 0; s < 2; ++s) {
            const float4 f0 = fA[s * 4 + 0], f1 = fA[s * 4 + 1];
            const float4 f2 = fA[s * 4 + 2], f3 = fA[s * 4 + 3];
            v16bf a;
            a[0]  = (__bf16)f0.x; a[1]  = (__bf16)f0.y;
            a[2]  = (__bf16)f0.z; a[3]  = (__bf16)f0.w;
            a[4]  = (__bf16)f1.x; a[5]  = (__bf16)f1.y;
            a[6]  = (__bf16)f1.z; a[7]  = (__bf16)f1.w;
            a[8]  = (__bf16)f2.x; a[9]  = (__bf16)f2.y;
            a[10] = (__bf16)f2.z; a[11] = (__bf16)f2.w;
            a[12] = (__bf16)f3.x; a[13] = (__bf16)f3.y;
            a[14] = (__bf16)f3.z; a[15] = (__bf16)f3.w;

            const int kkoff = s * 32 + lhalf * 16;
            v16bf bcur = *(const v16bf*)&Bt[buf][l15][kkoff];
            #pragma unroll
            for (int nt = 0; nt < 8; ++nt) {
                v16bf bnxt = bcur;
                if (nt < 7)
                    bnxt = *(const v16bf*)&Bt[buf][(nt + 1) * 16 + l15][kkoff];
                acc[nt] = __builtin_amdgcn_wmma_f32_16x16x32_bf16(
                    false, a, false, bcur, (short)0, acc[nt], false, false);
                bcur = bnxt;
            }
        }
        __syncthreads();
    }

    #pragma unroll
    for (int nt = 0; nt < 8; ++nt) {
        const int   n      = nt * 16 + l15;
        const float thresh = 0.5f - bias[n];
        #pragma unroll
        for (int v = 0; v < 8; ++v) {
            int row = mbase + v + lhalf * 8;
            if (row < rows)
                out[(size_t)row * HASH + n] = (acc[nt][v] > thresh) ? (signed char)1
                                                                    : (signed char)-1;
        }
    }
}

__global__ __launch_bounds__(256)
void hamming_argmin_kernel(const signed char* __restrict__ hf,
                           const signed char* __restrict__ hm,
                           unsigned long long* __restrict__ keys,
                           int mrows)
{
    __shared__ unsigned long long lkeys[64];
    const int tid   = threadIdx.x;
    const int wave  = tid >> 5;
    const int lane  = tid & 31;
    const int lhalf = lane >> 4;
    const int l15   = lane & 15;

    if (tid < 64) lkeys[tid] = ~0ull;
    __syncthreads();

    const int nrow = blockIdx.x * 128 + wave * 16 + l15;
    const int nclk = (nrow < mrows) ? nrow : (mrows - 1);
    const signed char* bptr = hm + (size_t)nclk * HASH;

    v8i acc[4] = {};

    #pragma unroll
    for (int c = 0; c < 2; ++c) {
        v8i b;
        {
            const int4 x0 = *(const int4*)(bptr + c * 64 + lhalf * 16);
            const int4 x1 = *(const int4*)(bptr + c * 64 + 32 + lhalf * 16);
            b[0] = x0.x; b[1] = x0.y; b[2] = x0.z; b[3] = x0.w;
            b[4] = x1.x; b[5] = x1.y; b[6] = x1.z; b[7] = x1.w;
        }
        #pragma unroll
        for (int m = 0; m < 4; ++m) {
            const signed char* ap = hf + (size_t)(m * 16 + l15) * HASH;
            v8i a;
            #pragma unroll
            for (int g = 0; g < 4; ++g) {
                const int* q = (const int*)(ap + c * 64 + (g >> 1) * 32 + (g & 1) * 8 + lhalf * 16);
                a[2 * g]     = q[0];
                a[2 * g + 1] = q[1];
            }
            acc[m] = __builtin_amdgcn_wmma_i32_16x16x64_iu8(
                true, a, true, b, acc[m], false, false);
        }
    }

    if (nrow < mrows) {
        #pragma unroll
        for (int m = 0; m < 4; ++m) {
            #pragma unroll
            for (int v = 0; v < 8; ++v) {
                const int fr  = m * 16 + v + lhalf * 8;
                const int dot = acc[m][v];
                unsigned long long key =
                    ((unsigned long long)(unsigned)(128 - dot) << 32) | (unsigned)nrow;
                atomicMin(&lkeys[fr], key);
            }
        }
    }
    __syncthreads();
    if (tid < 64) atomicMin(&keys[tid], lkeys[tid]);
}

__global__ void init_keys_kernel(unsigned long long* keys)
{
    keys[threadIdx.x] = ~0ull;
}

__global__ void gather_kernel(const unsigned long long* __restrict__ keys,
                              const float* __restrict__ memory,
                              float* __restrict__ out)
{
    const int b   = blockIdx.x;
    const int idx = (int)(keys[b] & 0xFFFFFFFFull);
    const float4* src = (const float4*)(memory + (size_t)idx * FEAT);
    float4*       dst = (float4*)(out + (size_t)b * FEAT);
    for (int i = threadIdx.x; i < FEAT / 4; i += blockDim.x) dst[i] = src[i];
}

extern "C" void kernel_launch(void* const* d_in, const int* in_sizes, int n_in,
                              void* d_out, int out_size, void* d_ws, size_t ws_size,
                              hipStream_t stream) {
    const float* feature = (const float*)d_in[0];
    const float* memory  = (const float*)d_in[1];
    const float* hash_W  = (const float*)d_in[2];
    const float* hash_b  = (const float*)d_in[3];
    float*       out     = (float*)d_out;

    const int M = 50000, B = 64;

    signed char* hm = (signed char*)d_ws;
    signed char* hf = hm + (size_t)M * HASH;
    unsigned long long* keys =
        (unsigned long long*)(hf + (size_t)B * HASH);
    __bf16* Wbf = (__bf16*)((char*)keys + 1024);

    const int mem_blocks = (M + TILE_M - 1) / TILE_M;

    convert_w_kernel<<<(HASH * FEAT / 8) / 256, 256, 0, stream>>>(hash_W, Wbf);
    binarize_wmma_kernel<<<mem_blocks, 256, 0, stream>>>(memory, Wbf, hash_b, hm, M);
    binarize_wmma_kernel<<<1,          256, 0, stream>>>(feature, Wbf, hash_b, hf, B);
    init_keys_kernel<<<1, 64, 0, stream>>>(keys);
    hamming_argmin_kernel<<<mem_blocks, 256, 0, stream>>>(hf, hm, keys, M);
    gather_kernel<<<B, 256, 0, stream>>>(keys, memory, out);
}
